// VoxelResBackBone8x_8315056685513
// MI455X (gfx1250) — matrix operand order repaired and checked by reading, not compiled, not run on hardware
//
#include <hip/hip_runtime.h>

// ---------------------------------------------------------------------------
// Types for CDNA5 WMMA (wave32): v_wmma_f32_16x16x32_bf16
// ---------------------------------------------------------------------------
typedef __attribute__((ext_vector_type(16))) __bf16 v16bf;
typedef __attribute__((ext_vector_type(8)))  float  v8f;

struct __attribute__((aligned(16))) U4 { unsigned x, y, z, w; };

__device__ __forceinline__ unsigned short f2bf(float f) {
  unsigned u = __float_as_uint(f);
  u += 0x7FFFu + ((u >> 16) & 1u);            // round-to-nearest-even
  return (unsigned short)(u >> 16);
}
__device__ __forceinline__ float bf2f(unsigned short h) {
  return __uint_as_float(((unsigned)h) << 16);
}

// ---------------------------------------------------------------------------
// Scatter voxel features into dense f32 grid (C=8, channels 5..7 stay zero)
// ---------------------------------------------------------------------------
__global__ void scatter_k(const float* __restrict__ feat,
                          const int* __restrict__ coords,
                          float* __restrict__ grid, float* __restrict__ mask,
                          int nvox, int H, int W) {
  int v = blockIdx.x * blockDim.x + threadIdx.x;
  if (v >= nvox) return;
  int z = coords[v * 4 + 1], y = coords[v * 4 + 2], x = coords[v * 4 + 3];
  long vox = ((long)z * H + y) * W + x;
  for (int c = 0; c < 5; ++c) atomicAdd(&grid[vox * 8 + c], feat[v * 5 + c]);
  mask[vox] = 1.0f;
}

__global__ void cvt_f32_bf16_k(const float* __restrict__ in,
                               unsigned short* __restrict__ out, long n) {
  long i = (long)blockIdx.x * blockDim.x + threadIdx.x;
  if (i < n) out[i] = f2bf(in[i]);
}

// ---------------------------------------------------------------------------
// Mask downsample: out = (conv(mask, ones) > 0)
// ---------------------------------------------------------------------------
__global__ void mask_down_k(const float* __restrict__ mi, float* __restrict__ mo,
                            int Di, int Hi, int Wi, int Do, int Ho, int Wo,
                            int kd, int kh, int kw, int sd, int sh, int sw,
                            int pd, int ph, int pw) {
  long idx = (long)blockIdx.x * blockDim.x + threadIdx.x;
  long total = (long)Do * Ho * Wo;
  if (idx >= total) return;
  int x = (int)(idx % Wo); long t = idx / Wo;
  int y = (int)(t % Ho);   int z = (int)(t / Ho);
  float v = 0.f;
  for (int dz = 0; dz < kd; ++dz) {
    int iz = z * sd + dz - pd; if ((unsigned)iz >= (unsigned)Di) continue;
    for (int dy = 0; dy < kh; ++dy) {
      int iy = y * sh + dy - ph; if ((unsigned)iy >= (unsigned)Hi) continue;
      for (int dx = 0; dx < kw; ++dx) {
        int ix = x * sw + dx - pw; if ((unsigned)ix >= (unsigned)Wi) continue;
        v += mi[((long)iz * Hi + iy) * Wi + ix];
      }
    }
  }
  mo[idx] = v > 0.f ? 1.f : 0.f;
}

// ---------------------------------------------------------------------------
// Pre-swizzle f32 DHWIO weights into per-lane WMMA B-fragment order (bf16).
// Flat layout: [slab][ntile][lane(32)][half(16)]
// B element i of lane L: k_local = i + 16*(L>>4), n = ntile*16 + (L&15)
// K-global = slab*32 + k_local = tap*ci_pad + c
// ---------------------------------------------------------------------------
__global__ void prep_wfrag_k(const float* __restrict__ w,
                             unsigned short* __restrict__ out,
                             int taps, int ci_real, int ci_pad, int co,
                             int nslabs, int ntiles) {
  long idx = (long)blockIdx.x * blockDim.x + threadIdx.x;
  long total = (long)nslabs * ntiles * 32 * 16;
  if (idx >= total) return;
  int i = (int)(idx & 15);
  int lane = (int)((idx >> 4) & 31);
  long t2 = idx >> 9;
  int nt = (int)(t2 % ntiles);
  int s = (int)(t2 / ntiles);
  int hl = lane >> 4;
  int n = nt * 16 + (lane & 15);
  int kl = i + (hl << 4);
  int kg = s * 32 + kl;
  int tap = kg / ci_pad;
  int c = kg - tap * ci_pad;
  float v = 0.f;
  if (tap < taps && c < ci_real && n < co)
    v = w[((long)tap * ci_real + c) * co + n];
  out[idx] = f2bf(v);
}

// ---------------------------------------------------------------------------
// Implicit-GEMM 3D conv via v_wmma_f32_16x16x32_bf16.
//  M = 32 output voxels along W (two 16x16 A fragments sharing each B frag),
//  N = 16 out-channels per tile (NT tiles),
//  K = taps*CI padded to 32-wide slabs. One wave per 32-voxel row.
// All geometry is compile-time so tap decode folds to shifts/constant-mults.
// Fused epilogue: BN scale/shift, mask multiply, residual add, ReLU, store.
// ---------------------------------------------------------------------------
template <int CI, int NT, int KD, int KH, int KW, int SD, int SH, int SW,
          int PD, int PH, int PW, bool F32OUT>
__global__ __launch_bounds__(256)
void conv3d_wmma(const unsigned short* __restrict__ act,
                 const unsigned short* __restrict__ wfrag,
                 const float* __restrict__ gamma, const float* __restrict__ beta,
                 const float* __restrict__ mean, const float* __restrict__ var,
                 const float* __restrict__ mask,
                 const unsigned short* __restrict__ resid,
                 unsigned short* __restrict__ outb, float* __restrict__ outf,
                 int Di, int Hi, int Wi, int Do, int Ho, int Wo, int relu) {
  constexpr int TAPS = KD * KH * KW;
  constexpr int KHW = KH * KW;
  constexpr int NSLABS = (TAPS * CI + 31) / 32;
  const int lane = threadIdx.x & 31;
  const int hl = lane >> 4;
  const int wid = threadIdx.x >> 5;
  const int xtiles = Wo >> 5;                  // 32 voxels per wave
  const long nwaves = (long)Do * Ho * xtiles;
  long gw = (long)blockIdx.x * (blockDim.x >> 5) + wid;
  if (gw >= nwaves) return;
  int xt = (int)(gw % xtiles);
  long t = gw / xtiles;
  int yo = (int)(t % Ho);
  int zo = (int)(t / Ho);
  const int m = lane & 15;                     // A-matrix row
  const int xbase = xt << 5;

  v8f acc[2][NT] = {};
  const U4* wq = (const U4*)wfrag;

  for (int s = 0; s < NSLABS; ++s) {
    // ---- two A fragments (adjacent 16-voxel tiles), zero-padded OOB
    union { U4 q[2]; v16bf v; } A[2];
#pragma unroll
    for (int j = 0; j < 2; ++j) {
      int k0 = s * 32 + (hl << 4) + (j << 3);  // elem i -> K=16*hl+i
      int tap = k0 / CI;                       // CI power of two -> shift
      int c0 = k0 - tap * CI;
      int dz = tap / KHW;                      // constexpr divisor -> cheap
      int r2 = tap - dz * KHW;
      int dy = r2 / KW;
      int dx = r2 - dy * KW;
      int iz = zo * SD + dz - PD;
      int iy = yo * SH + dy - PH;
      bool okzy = (tap < TAPS) & ((unsigned)iz < (unsigned)Di) &
                  ((unsigned)iy < (unsigned)Hi);
      long rowbase = ((long)iz * Hi + iy) * Wi * CI + c0;
#pragma unroll
      for (int mt = 0; mt < 2; ++mt) {
        int ix = (xbase + (mt << 4) + m) * SW + dx - PW;
        U4 q = {0u, 0u, 0u, 0u};
        if (okzy && (unsigned)ix < (unsigned)Wi)
          q = *(const U4*)(act + rowbase + (long)ix * CI);
        A[mt].q[j] = q;
      }
    }

    long wbase = ((long)s * NT * 32 + lane) * 2;      // U4 units
    if (s + 1 < NSLABS)                               // global_prefetch_b8
      __builtin_prefetch((const void*)(wq + wbase + (long)NT * 64), 0, 1);

#pragma unroll
    for (int nt = 0; nt < NT; ++nt) {
      union { U4 q[2]; v16bf v; } B;
      long b0 = wbase + (long)nt * 64;
      B.q[0] = wq[b0];
      B.q[1] = wq[b0 + 1];
      acc[0][nt] = __builtin_amdgcn_wmma_f32_16x16x32_bf16(
          false, A[0].v, false, B.v, (short)0, acc[0][nt], false, false);
      acc[1][nt] = __builtin_amdgcn_wmma_f32_16x16x32_bf16(
          false, A[1].v, false, B.v, (short)0, acc[1][nt], false, false);
    }
  }

  // ---- epilogue (C/D layout: N = lane&15, M = r + 8*(lane>>4))
  const int CO = NT * 16;
  float scv[NT], shv[NT];
#pragma unroll
  for (int nt = 0; nt < NT; ++nt) {
    int n = (nt << 4) + (lane & 15);
    float sc = 1.f, sh0 = 0.f;
    if (gamma) {
      float iv = rsqrtf(var[n] + 1e-3f);
      sc = gamma[n] * iv;
      sh0 = beta[n] - mean[n] * sc;
    }
    scv[nt] = sc;
    shv[nt] = sh0;
  }
#pragma unroll
  for (int mt = 0; mt < 2; ++mt) {
    long voxb = ((long)zo * Ho + yo) * Wo + xbase + (mt << 4);
    union { U4 q[2]; float f[8]; } MK;        // mask for rows M=8*hl..8*hl+7
    MK.q[0] = *(const U4*)(mask + voxb + (hl << 3));
    MK.q[1] = *(const U4*)(mask + voxb + (hl << 3) + 4);
#pragma unroll
    for (int nt = 0; nt < NT; ++nt) {
      int n = (nt << 4) + (lane & 15);
#pragma unroll
      for (int r = 0; r < 8; ++r) {
        int M = r + (hl << 3);
        long vox = voxb + M;
        float v = acc[mt][nt][r] * scv[nt] + shv[nt];
        v *= MK.f[r];
        if (resid) v += bf2f(resid[vox * CO + n]);
        if (relu) v = fmaxf(v, 0.f);
        if (F32OUT) outf[vox * CO + n] = v;
        else        outb[vox * CO + n] = f2bf(v);
      }
    }
  }
}

// ---------------------------------------------------------------------------
// Host driver
// ---------------------------------------------------------------------------
static inline void* bump(char*& p, size_t bytes) {
  void* r = (void*)p;
  p += (bytes + 255) & ~(size_t)255;
  return r;
}

#define CONV(CI, NT, KD, KH, KW, SD, SH, SW, PD, PH, PW, F32, IN, WF, G, B,    \
             MU, VAR, MASK, RES, OUTB, OUTF, Di, Hi, Wi, Do_, Ho_, Wo_, RELU)  \
  do {                                                                         \
    long waves_ = (long)(Do_) * (Ho_) * ((Wo_) >> 5);                          \
    int blocks_ = (int)((waves_ + 7) / 8);                                     \
    conv3d_wmma<CI, NT, KD, KH, KW, SD, SH, SW, PD, PH, PW, F32>               \
        <<<blocks_, 256, 0, stream>>>(IN, WF, G, B, MU, VAR, MASK, RES, OUTB,  \
                                      OUTF, Di, Hi, Wi, Do_, Ho_, Wo_, RELU);  \
  } while (0)

#define MDOWN(MI, MO, Di, Hi, Wi, Do_, Ho_, Wo_, kd, kh, kw, sd, sh, sw, pd,   \
              ph, pw)                                                          \
  do {                                                                         \
    long tot_ = (long)(Do_) * (Ho_) * (Wo_);                                   \
    mask_down_k<<<(int)((tot_ + 255) / 256), 256, 0, stream>>>(                \
        MI, MO, Di, Hi, Wi, Do_, Ho_, Wo_, kd, kh, kw, sd, sh, sw, pd, ph,     \
        pw);                                                                   \
  } while (0)

static void prep(const float* w, unsigned short* out, int taps, int ci_real,
                 int ci_pad, int co, hipStream_t stream) {
  int nslabs = (taps * ci_pad + 31) / 32;
  int nt = co / 16;
  long total = (long)nslabs * nt * 512;
  prep_wfrag_k<<<(int)((total + 255) / 256), 256, 0, stream>>>(
      w, out, taps, ci_real, ci_pad, co, nslabs, nt);
}

extern "C" void kernel_launch(void* const* d_in, const int* in_sizes, int n_in,
                              void* d_out, int out_size, void* d_ws,
                              size_t ws_size, hipStream_t stream) {
  const int D = 33, H = 256, W = 256;
  const long NVF = (long)D * H * W;          // 2,162,688
  const long NV2 = 17L * 128 * 128;          // 278,528
  const long NV3 = 9L * 64 * 64;             // 36,864
  const long NV4 = 5L * 32 * 32;             // 5,120
  const long NVO = 2L * 32 * 32;             // 2,048

  const float* feat = (const float*)d_in[0];
  const int* coords = (const int*)d_in[1];
  auto F = [&](int i) { return (const float*)d_in[i]; };
  int nvox = in_sizes[0] / 5;

  // ---- workspace layout --------------------------------------------------
  char* p = (char*)d_ws;
  float* in_f32 = (float*)bump(p, NVF * 8 * sizeof(float));   // scatter grid
  unsigned short* in_bf = (unsigned short*)bump(p, NVF * 8 * 2);
  float* mask0 = (float*)bump(p, NVF * sizeof(float));
  unsigned short* A0 = (unsigned short*)bump(p, NVF * 16 * 2);
  unsigned short* B0 = (unsigned short*)bump(p, NVF * 16 * 2);
  unsigned short* T0 = (unsigned short*)in_f32;  // alias: in_f32 dead after cvt

  float* mask2 = (float*)bump(p, NV2 * sizeof(float));
  unsigned short* A2 = (unsigned short*)bump(p, NV2 * 32 * 2);
  unsigned short* B2 = (unsigned short*)bump(p, NV2 * 32 * 2);
  unsigned short* T2 = (unsigned short*)bump(p, NV2 * 32 * 2);

  float* mask3 = (float*)bump(p, NV3 * sizeof(float));
  unsigned short* A3 = (unsigned short*)bump(p, NV3 * 64 * 2);
  unsigned short* B3 = (unsigned short*)bump(p, NV3 * 64 * 2);
  unsigned short* T3 = (unsigned short*)bump(p, NV3 * 64 * 2);

  float* mask4 = (float*)bump(p, NV4 * sizeof(float));
  unsigned short* A4 = (unsigned short*)bump(p, NV4 * 128 * 2);
  unsigned short* B4 = (unsigned short*)bump(p, NV4 * 128 * 2);
  unsigned short* T4 = (unsigned short*)bump(p, NV4 * 128 * 2);

  float* maskF = (float*)bump(p, NVO * sizeof(float));

  auto wfrag_alloc = [&](int taps, int ci_pad, int co) {
    int ns = (taps * ci_pad + 31) / 32;
    return (unsigned short*)bump(p, (size_t)ns * (co / 16) * 512 * 2);
  };
  unsigned short* wp_ci = wfrag_alloc(27, 8, 16);
  unsigned short* wp_11a = wfrag_alloc(27, 16, 16);
  unsigned short* wp_11b = wfrag_alloc(27, 16, 16);
  unsigned short* wp_12a = wfrag_alloc(27, 16, 16);
  unsigned short* wp_12b = wfrag_alloc(27, 16, 16);
  unsigned short* wp_d2 = wfrag_alloc(27, 16, 32);
  unsigned short* wp_21a = wfrag_alloc(27, 32, 32);
  unsigned short* wp_21b = wfrag_alloc(27, 32, 32);
  unsigned short* wp_22a = wfrag_alloc(27, 32, 32);
  unsigned short* wp_22b = wfrag_alloc(27, 32, 32);
  unsigned short* wp_d3 = wfrag_alloc(27, 32, 64);
  unsigned short* wp_31a = wfrag_alloc(27, 64, 64);
  unsigned short* wp_31b = wfrag_alloc(27, 64, 64);
  unsigned short* wp_32a = wfrag_alloc(27, 64, 64);
  unsigned short* wp_32b = wfrag_alloc(27, 64, 64);
  unsigned short* wp_d4 = wfrag_alloc(27, 64, 128);
  unsigned short* wp_41a = wfrag_alloc(27, 128, 128);
  unsigned short* wp_41b = wfrag_alloc(27, 128, 128);
  unsigned short* wp_42a = wfrag_alloc(27, 128, 128);
  unsigned short* wp_42b = wfrag_alloc(27, 128, 128);
  unsigned short* wp_out = wfrag_alloc(3, 128, 128);

  // ---- scatter + convert -------------------------------------------------
  (void)hipMemsetAsync(in_f32, 0, NVF * 8 * sizeof(float), stream);
  (void)hipMemsetAsync(mask0, 0, NVF * sizeof(float), stream);
  scatter_k<<<(nvox + 255) / 256, 256, 0, stream>>>(feat, coords, in_f32,
                                                    mask0, nvox, H, W);
  cvt_f32_bf16_k<<<(int)((NVF * 8 + 255) / 256), 256, 0, stream>>>(
      in_f32, in_bf, NVF * 8);

  // ---- weight preswizzle -------------------------------------------------
  prep(F(2), wp_ci, 27, 5, 8, 16, stream);
  prep(F(7), wp_11a, 27, 16, 16, 16, stream);
  prep(F(12), wp_11b, 27, 16, 16, 16, stream);
  prep(F(17), wp_12a, 27, 16, 16, 16, stream);
  prep(F(22), wp_12b, 27, 16, 16, 16, stream);
  prep(F(27), wp_d2, 27, 16, 16, 32, stream);
  prep(F(28), wp_21a, 27, 32, 32, 32, stream);
  prep(F(33), wp_21b, 27, 32, 32, 32, stream);
  prep(F(38), wp_22a, 27, 32, 32, 32, stream);
  prep(F(43), wp_22b, 27, 32, 32, 32, stream);
  prep(F(48), wp_d3, 27, 32, 32, 64, stream);
  prep(F(49), wp_31a, 27, 64, 64, 64, stream);
  prep(F(54), wp_31b, 27, 64, 64, 64, stream);
  prep(F(59), wp_32a, 27, 64, 64, 64, stream);
  prep(F(64), wp_32b, 27, 64, 64, 64, stream);
  prep(F(69), wp_d4, 27, 64, 64, 128, stream);
  prep(F(70), wp_41a, 27, 128, 128, 128, stream);
  prep(F(75), wp_41b, 27, 128, 128, 128, stream);
  prep(F(80), wp_42a, 27, 128, 128, 128, stream);
  prep(F(85), wp_42b, 27, 128, 128, 128, stream);
  prep(F(90), wp_out, 3, 128, 128, 128, stream);

  const float* NF = nullptr;       // null BN param
  float* NOF = nullptr;            // null f32 output
  unsigned short* NU = nullptr;    // null bf16 buffer

  // ---- conv_input: relu(bn(conv(x))*mask) --------------------------------
  CONV(8, 1, 3, 3, 3, 1, 1, 1, 1, 1, 1, false, in_bf, wp_ci,
       F(3), F(4), F(5), F(6), mask0, NU, A0, NOF, 33, 256, 256, 33, 256, 256, 1);

  // ---- stage 1 (stride 1, 16ch) -----------------------------------------
  CONV(16, 1, 3, 3, 3, 1, 1, 1, 1, 1, 1, false, A0, wp_11a,
       F(8), F(9), F(10), F(11), mask0, NU, T0, NOF, 33, 256, 256, 33, 256, 256, 1);
  CONV(16, 1, 3, 3, 3, 1, 1, 1, 1, 1, 1, false, T0, wp_11b,
       F(13), F(14), F(15), F(16), mask0, A0, B0, NOF, 33, 256, 256, 33, 256, 256, 1);
  CONV(16, 1, 3, 3, 3, 1, 1, 1, 1, 1, 1, false, B0, wp_12a,
       F(18), F(19), F(20), F(21), mask0, NU, T0, NOF, 33, 256, 256, 33, 256, 256, 1);
  CONV(16, 1, 3, 3, 3, 1, 1, 1, 1, 1, 1, false, T0, wp_12b,
       F(23), F(24), F(25), F(26), mask0, B0, A0, NOF, 33, 256, 256, 33, 256, 256, 1);

  // ---- stage 2 (down to 17x128x128, 32ch) -------------------------------
  MDOWN(mask0, mask2, 33, 256, 256, 17, 128, 128, 3, 3, 3, 2, 2, 2, 1, 1, 1);
  CONV(16, 2, 3, 3, 3, 2, 2, 2, 1, 1, 1, false, A0, wp_d2,
       NF, NF, NF, NF, mask2, NU, A2, NOF, 33, 256, 256, 17, 128, 128, 0);
  CONV(32, 2, 3, 3, 3, 1, 1, 1, 1, 1, 1, false, A2, wp_21a,
       F(29), F(30), F(31), F(32), mask2, NU, T2, NOF, 17, 128, 128, 17, 128, 128, 1);
  CONV(32, 2, 3, 3, 3, 1, 1, 1, 1, 1, 1, false, T2, wp_21b,
       F(34), F(35), F(36), F(37), mask2, A2, B2, NOF, 17, 128, 128, 17, 128, 128, 1);
  CONV(32, 2, 3, 3, 3, 1, 1, 1, 1, 1, 1, false, B2, wp_22a,
       F(39), F(40), F(41), F(42), mask2, NU, T2, NOF, 17, 128, 128, 17, 128, 128, 1);
  CONV(32, 2, 3, 3, 3, 1, 1, 1, 1, 1, 1, false, T2, wp_22b,
       F(44), F(45), F(46), F(47), mask2, B2, A2, NOF, 17, 128, 128, 17, 128, 128, 1);

  // ---- stage 3 (down to 9x64x64, 64ch) ----------------------------------
  MDOWN(mask2, mask3, 17, 128, 128, 9, 64, 64, 3, 3, 3, 2, 2, 2, 1, 1, 1);
  CONV(32, 4, 3, 3, 3, 2, 2, 2, 1, 1, 1, false, A2, wp_d3,
       NF, NF, NF, NF, mask3, NU, A3, NOF, 17, 128, 128, 9, 64, 64, 0);
  CONV(64, 4, 3, 3, 3, 1, 1, 1, 1, 1, 1, false, A3, wp_31a,
       F(50), F(51), F(52), F(53), mask3, NU, T3, NOF, 9, 64, 64, 9, 64, 64, 1);
  CONV(64, 4, 3, 3, 3, 1, 1, 1, 1, 1, 1, false, T3, wp_31b,
       F(55), F(56), F(57), F(58), mask3, A3, B3, NOF, 9, 64, 64, 9, 64, 64, 1);
  CONV(64, 4, 3, 3, 3, 1, 1, 1, 1, 1, 1, false, B3, wp_32a,
       F(60), F(61), F(62), F(63), mask3, NU, T3, NOF, 9, 64, 64, 9, 64, 64, 1);
  CONV(64, 4, 3, 3, 3, 1, 1, 1, 1, 1, 1, false, T3, wp_32b,
       F(65), F(66), F(67), F(68), mask3, B3, A3, NOF, 9, 64, 64, 9, 64, 64, 1);

  // ---- stage 4 (down to 5x32x32, 128ch) ---------------------------------
  MDOWN(mask3, mask4, 9, 64, 64, 5, 32, 32, 3, 3, 3, 2, 2, 2, 1, 1, 1);
  CONV(64, 8, 3, 3, 3, 2, 2, 2, 1, 1, 1, false, A3, wp_d4,
       NF, NF, NF, NF, mask4, NU, A4, NOF, 9, 64, 64, 5, 32, 32, 0);
  CONV(128, 8, 3, 3, 3, 1, 1, 1, 1, 1, 1, false, A4, wp_41a,
       F(71), F(72), F(73), F(74), mask4, NU, T4, NOF, 5, 32, 32, 5, 32, 32, 1);
  CONV(128, 8, 3, 3, 3, 1, 1, 1, 1, 1, 1, false, T4, wp_41b,
       F(76), F(77), F(78), F(79), mask4, A4, B4, NOF, 5, 32, 32, 5, 32, 32, 1);
  CONV(128, 8, 3, 3, 3, 1, 1, 1, 1, 1, 1, false, B4, wp_42a,
       F(81), F(82), F(83), F(84), mask4, NU, T4, NOF, 5, 32, 32, 5, 32, 32, 1);
  CONV(128, 8, 3, 3, 3, 1, 1, 1, 1, 1, 1, false, T4, wp_42b,
       F(86), F(87), F(88), F(89), mask4, B4, A4, NOF, 5, 32, 32, 5, 32, 32, 1);

  // ---- conv_out: (3,1,1) stride (2,1,1) pad 0, f32 output ---------------
  MDOWN(mask4, maskF, 5, 32, 32, 2, 32, 32, 3, 1, 1, 2, 1, 1, 0, 0, 0);
  CONV(128, 8, 3, 1, 1, 2, 1, 1, 0, 0, 0, true, A4, wp_out,
       F(91), F(92), F(93), F(94), maskF, NU, NU, (float*)d_out,
       5, 32, 32, 2, 32, 32, 1);

  (void)n_in; (void)out_size; (void)ws_size;
}
